// GAT_GCN_49117245997773
// MI455X (gfx1250) — compile-verified, matrix operand order checked by reading, not run on hardware
//
#include <hip/hip_runtime.h>
#include <hip/hip_bf16.h>
#include <math.h>

typedef __attribute__((ext_vector_type(16))) __bf16 v16bf;
typedef __attribute__((ext_vector_type(8)))  __bf16 v8bf;
typedef __attribute__((ext_vector_type(2)))  __bf16 v2bf;
typedef __attribute__((ext_vector_type(8)))  float  v8f;

#define NNODES 40000
#define NEDGES 160000
#define ETOT   200000   // edges + self loops
#define NB     1024
#define FXD    78
#define NH     10
#define HF     780      // NH*FXD
#define DOUT   128
#define DP     256

#define ORD_NEG_INF 0x007FFFFFu   // f2o(-inf)

// ---- order-preserving float<->uint map (for atomic max on floats) ----
__device__ __forceinline__ unsigned f2o(float f) {
  unsigned u = __float_as_uint(f);
  return (u & 0x80000000u) ? ~u : (u | 0x80000000u);
}
__device__ __forceinline__ float o2f(unsigned u) {
  return (u & 0x80000000u) ? __uint_as_float(u & 0x7fffffffu) : __uint_as_float(~u);
}

// =====================================================================
// WMMA GEMM: C[M,N] = act(A[M,K] @ W[K,N] (+ bias))
// Requirements: M % 64 == 0 and K even (true at all call sites).
// f32 in HBM -> packed bf16 in LDS -> v_wmma_f32_16x16x32_bf16.
// Block = 256 threads (8 wave32). Block tile 64(M) x 128(N), K staged
// 64 per barrier; K tail peeled so the steady-state loop is guard-free.
// Wave tile 32x32 = 4 accumulators -> 4 wmma per 8 ds_load_b128.
// LDS rows padded to 72 bf16 (144B: 16B-aligned, bank-spread).
// A stored [m][k]; B stored TRANSPOSED [n][k] so all fragments are two
// contiguous 8xbf16 runs.
// flags: bit0 = relu, bit1 = add bias
// =====================================================================
__global__ __launch_bounds__(256) void wmma_gemm_k(
    const float* __restrict__ A, const float* __restrict__ W,
    const float* __restrict__ bias, float* __restrict__ C,
    int M, int N, int K, int flags) {
  __shared__ __align__(16) __bf16 As[64][72];
  __shared__ __align__(16) __bf16 Bs[128][72];   // [n][k]
  const int tid  = threadIdx.x;
  const int lane = tid & 31;
  const int wid  = tid >> 5;
  const int wr   = wid >> 2;       // 0..1 : wave row group (32 rows)
  const int wc   = wid & 3;        // 0..3 : wave col group (32 cols)
  const int bm   = blockIdx.y * 64;
  const int bn   = blockIdx.x * 128;
  const int lm   = lane & 15;
  const int lh   = lane >> 4;      // lane half

  // loop-invariant staging coordinates
  // A: unit i -> row ra[i] (32 pairs/row), pair col pa[i]
  // B: unit j (16) -> col nb[j] (clamped sb[j]), kpair kb[j], ok flag
  int ra[8], pa[8];
  #pragma unroll
  for (int i = 0; i < 8; ++i) {
    int p = i * 256 + tid;
    ra[i] = p >> 5;
    pa[i] = p & 31;
  }
  int nbv[16], kbv[16], sbv[16];
  bool okn[16];
  #pragma unroll
  for (int j = 0; j < 16; ++j) {
    int u = j * 256 + tid;
    nbv[j] = u & 127;
    kbv[j] = u >> 7;
    int gn = bn + nbv[j];
    okn[j] = (gn < N);
    sbv[j] = okn[j] ? gn : 0;
  }

  v8f acc00 = {}, acc01 = {}, acc10 = {}, acc11 = {};

  const __bf16* arow0 = &As[wr * 32 + lm][0];
  const __bf16* arow1 = &As[wr * 32 + 16 + lm][0];
  const __bf16* brow0 = &Bs[wc * 32 + lm][0];
  const __bf16* brow1 = &Bs[wc * 32 + 16 + lm][0];

  auto compute = [&]() {
    #pragma unroll
    for (int ks = 0; ks < 64; ks += 32) {
      union { v16bf v; v8bf h[2]; } a0, a1, b0, b1;
      a0.h[0] = *(const v8bf*)(arow0 + ks + lh * 8);        // ds_load_b128
      a0.h[1] = *(const v8bf*)(arow0 + ks + 16 + lh * 8);
      a1.h[0] = *(const v8bf*)(arow1 + ks + lh * 8);
      a1.h[1] = *(const v8bf*)(arow1 + ks + 16 + lh * 8);
      b0.h[0] = *(const v8bf*)(brow0 + ks + lh * 8);
      b0.h[1] = *(const v8bf*)(brow0 + ks + 16 + lh * 8);
      b1.h[0] = *(const v8bf*)(brow1 + ks + lh * 8);
      b1.h[1] = *(const v8bf*)(brow1 + ks + 16 + lh * 8);
      acc00 = __builtin_amdgcn_wmma_f32_16x16x32_bf16(false, a0.v, false, b0.v, (short)0, acc00, false, false);
      acc01 = __builtin_amdgcn_wmma_f32_16x16x32_bf16(false, a0.v, false, b1.v, (short)0, acc01, false, false);
      acc10 = __builtin_amdgcn_wmma_f32_16x16x32_bf16(false, a1.v, false, b0.v, (short)0, acc10, false, false);
      acc11 = __builtin_amdgcn_wmma_f32_16x16x32_bf16(false, a1.v, false, b1.v, (short)0, acc11, false, false);
    }
  };

  const int kfull = K & ~63;

  // ---------------- steady state: no K guards at all ----------------
  for (int k0 = 0; k0 < kfull; k0 += 64) {
    {
      float2 ld[8];
      #pragma unroll
      for (int i = 0; i < 8; ++i)
        ld[i] = *(const float2*)(A + (size_t)(bm + ra[i]) * K + k0 + pa[i] * 2);
      #pragma unroll
      for (int i = 0; i < 8; ++i) {
        v2bf t; t[0] = (__bf16)ld[i].x; t[1] = (__bf16)ld[i].y;
        *(v2bf*)&As[ra[i]][pa[i] * 2] = t;
      }
    }
    #pragma unroll
    for (int half = 0; half < 2; ++half) {
      float l0[8], l1[8];
      #pragma unroll
      for (int i = 0; i < 8; ++i) {
        int j = half * 8 + i;
        int gk = k0 + kbv[j] * 2;
        l0[i] = W[(size_t)gk * N + sbv[j]];
        l1[i] = W[(size_t)(gk + 1) * N + sbv[j]];
      }
      #pragma unroll
      for (int i = 0; i < 8; ++i) {
        int j = half * 8 + i;
        float vx = okn[j] ? l0[i] : 0.0f;   // N-edge value select only
        float vy = okn[j] ? l1[i] : 0.0f;
        v2bf t; t[0] = (__bf16)vx; t[1] = (__bf16)vy;
        *(v2bf*)&Bs[nbv[j]][kbv[j] * 2] = t;
      }
    }
    __syncthreads();
    compute();
    __syncthreads();
  }

  // ---------------- peeled K tail (guarded; K even -> pair all-or-nothing) ----
  if (kfull < K) {
    const int k0 = kfull;
    {
      float2 ld[8];
      bool   ok[8];
      #pragma unroll
      for (int i = 0; i < 8; ++i) {
        int gk = k0 + pa[i] * 2;
        ok[i] = (gk < K);
        int sk = ok[i] ? gk : 0;
        ld[i] = *(const float2*)(A + (size_t)(bm + ra[i]) * K + sk);
      }
      #pragma unroll
      for (int i = 0; i < 8; ++i) {
        float vx = ok[i] ? ld[i].x : 0.0f;
        float vy = ok[i] ? ld[i].y : 0.0f;
        v2bf t; t[0] = (__bf16)vx; t[1] = (__bf16)vy;
        *(v2bf*)&As[ra[i]][pa[i] * 2] = t;
      }
    }
    #pragma unroll
    for (int half = 0; half < 2; ++half) {
      float l0[8], l1[8];
      bool  ok[8];
      #pragma unroll
      for (int i = 0; i < 8; ++i) {
        int j = half * 8 + i;
        int gk = k0 + kbv[j] * 2;
        ok[i] = (gk < K);
        int sk = ok[i] ? gk : 0;
        l0[i] = W[(size_t)sk * N + sbv[j]];
        l1[i] = W[(size_t)(sk + (ok[i] ? 1 : 0)) * N + sbv[j]];
      }
      #pragma unroll
      for (int i = 0; i < 8; ++i) {
        int j = half * 8 + i;
        bool v = ok[i] && okn[j];
        float vx = v ? l0[i] : 0.0f;
        float vy = v ? l1[i] : 0.0f;
        v2bf t; t[0] = (__bf16)vx; t[1] = (__bf16)vy;
        *(v2bf*)&Bs[nbv[j]][kbv[j] * 2] = t;
      }
    }
    __syncthreads();
    compute();
  }

  // store C: VGPR r -> M = r + 8*lh (within 16-row tile); N = lane&15
  const int rowA0 = bm + wr * 32 + lh * 8;       // a0 tiles
  const int rowA1 = rowA0 + 16;                  // a1 tiles
  const int gn0   = bn + wc * 32 + lm;
  const int gn1   = gn0 + 16;
  float bias0 = 0.f, bias1 = 0.f;
  if (flags & 2) {
    if (gn0 < N) bias0 = bias[gn0];
    if (gn1 < N) bias1 = bias[gn1];
  }
  #pragma unroll
  for (int r = 0; r < 8; ++r) {
    int gm0 = rowA0 + r;
    int gm1 = rowA1 + r;
    if (gn0 < N) {
      float v = acc00[r] + bias0;
      if (flags & 1) v = fmaxf(v, 0.0f);
      C[(size_t)gm0 * N + gn0] = v;
      v = acc10[r] + bias0;
      if (flags & 1) v = fmaxf(v, 0.0f);
      C[(size_t)gm1 * N + gn0] = v;
    }
    if (gn1 < N) {
      float v = acc01[r] + bias1;
      if (flags & 1) v = fmaxf(v, 0.0f);
      C[(size_t)gm0 * N + gn1] = v;
      v = acc11[r] + bias1;
      if (flags & 1) v = fmaxf(v, 0.0f);
      C[(size_t)gm1 * N + gn1] = v;
    }
  }
}

// =====================================================================
// small utility / irregular kernels
// =====================================================================
__global__ void fill_f32_k(float* p, float v, int n) {
  int i = blockIdx.x * blockDim.x + threadIdx.x;
  if (i < n) p[i] = v;
}
__global__ void fill_u32_k(unsigned* p, unsigned v, int n) {
  int i = blockIdx.x * blockDim.x + threadIdx.x;
  if (i < n) p[i] = v;
}
__global__ void bias_act_k(float* p, const float* __restrict__ bias, int rows, int cols, int relu) {
  int i = blockIdx.x * blockDim.x + threadIdx.x;
  if (i >= rows * cols) return;
  float v = p[i] + bias[i % cols];
  p[i] = relu ? fmaxf(v, 0.0f) : v;
}

// per-(node,head) GAT attention logits: asrc/adst = <h[n,h,:], att>
__global__ void gat_scores_k(const float* __restrict__ hlin, const float* __restrict__ att_src,
                             const float* __restrict__ att_dst, float* __restrict__ asrc,
                             float* __restrict__ adst) {
  int i = blockIdx.x * blockDim.x + threadIdx.x;
  if (i >= NNODES * NH) return;
  int n = i / NH, h = i - n * NH;
  const float* hp = hlin + (size_t)n * HF + h * FXD;
  const float* as = att_src + h * FXD;
  const float* ad = att_dst + h * FXD;
  float s1 = 0.f, s2 = 0.f;
  for (int c = 0; c < FXD; ++c) { float v = hp[c]; s1 += v * as[c]; s2 += v * ad[c]; }
  asrc[i] = s1; adst[i] = s2;
}

// alpha = leakyrelu(asrc[src]+adst[dst]); segment max over dst (atomic, ordered)
__global__ void edge_alpha_max_k(const int* __restrict__ ei, const float* __restrict__ asrc,
                                 const float* __restrict__ adst, float* __restrict__ alpha,
                                 unsigned* __restrict__ mseg) {
  int i = blockIdx.x * blockDim.x + threadIdx.x;
  if (i >= ETOT * NH) return;
  int e = i / NH, h = i - e * NH;
  int s = (e < NEDGES) ? ei[e]          : (e - NEDGES);
  int d = (e < NEDGES) ? ei[NEDGES + e] : (e - NEDGES);
  float a = asrc[s * NH + h] + adst[d * NH + h];
  a = (a > 0.f) ? a : 0.2f * a;
  alpha[i] = a;
  atomicMax(&mseg[d * NH + h], f2o(a));
}

// e = exp(alpha - m[dst]); segment sum over dst
__global__ void edge_expsum_k(const int* __restrict__ ei, float* __restrict__ alpha,
                              const unsigned* __restrict__ mseg, float* __restrict__ sseg) {
  int i = blockIdx.x * blockDim.x + threadIdx.x;
  if (i >= ETOT * NH) return;
  int e = i / NH, h = i - e * NH;
  int d = (e < NEDGES) ? ei[NEDGES + e] : (e - NEDGES);
  float ev = __expf(alpha[i] - o2f(mseg[d * NH + h]));
  alpha[i] = ev;                  // reuse as e
  atomicAdd(&sseg[d * NH + h], ev);
}

// out[dst, :] += attn[h] * h[src, :]   (one block per edge, 780 feats)
__global__ void gat_aggregate_k(const int* __restrict__ ei, const float* __restrict__ ealpha,
                                const float* __restrict__ sseg, const float* __restrict__ hlin,
                                float* __restrict__ out) {
  int e = blockIdx.x;
  int s = (e < NEDGES) ? ei[e]          : (e - NEDGES);
  int d = (e < NEDGES) ? ei[NEDGES + e] : (e - NEDGES);
  __shared__ float attn[NH];
  if (threadIdx.x < NH)
    attn[threadIdx.x] = ealpha[e * NH + threadIdx.x] /
                        (sseg[d * NH + threadIdx.x] + 1e-16f);
  __syncthreads();
  const float* hp = hlin + (size_t)s * HF;
  float* op = out + (size_t)d * HF;
  for (int c = threadIdx.x; c < HF; c += blockDim.x)
    atomicAdd(&op[c], attn[c / FXD] * hp[c]);
}

__global__ void deg_k(const int* __restrict__ ei, float* __restrict__ deg) {
  int e = blockIdx.x * blockDim.x + threadIdx.x;
  if (e >= ETOT) return;
  int d = (e < NEDGES) ? ei[NEDGES + e] : (e - NEDGES);
  atomicAdd(&deg[d], 1.0f);
}
__global__ void dinv_k(const float* __restrict__ deg, float* __restrict__ dinv) {
  int i = blockIdx.x * blockDim.x + threadIdx.x;
  if (i >= NNODES) return;
  float d = deg[i];
  dinv[i] = (d > 0.f) ? rsqrtf(d) : 0.f;
}

// GCN aggregate: out[dst,:] += dinv[src]*dinv[dst] * hW[src,:]
__global__ void gcn_aggregate_k(const int* __restrict__ ei, const float* __restrict__ dinv,
                                const float* __restrict__ hw, float* __restrict__ out) {
  int e = blockIdx.x;
  int s = (e < NEDGES) ? ei[e]          : (e - NEDGES);
  int d = (e < NEDGES) ? ei[NEDGES + e] : (e - NEDGES);
  float norm = dinv[s] * dinv[d];
  const float* hp = hw + (size_t)s * HF;
  float* op = out + (size_t)d * HF;
  for (int c = threadIdx.x; c < HF; c += blockDim.x)
    atomicAdd(&op[c], norm * hp[c]);
}

// cross-attn scores per node against its graph's key
__global__ void cross_scores_k(const float* __restrict__ q, const float* __restrict__ kmat,
                               const int* __restrict__ batch, float* __restrict__ sc,
                               unsigned* __restrict__ mb) {
  int n = blockIdx.x * blockDim.x + threadIdx.x;
  if (n >= NNODES) return;
  int b = batch[n];
  const float* qp = q + (size_t)n * 128;
  const float* kp = kmat + (size_t)b * 128;
  float s = 0.f;
  for (int i = 0; i < 128; ++i) s += qp[i] * kp[i];
  s *= 0.08838834764831845f;   // 1/sqrt(128)
  sc[n] = s;
  atomicMax(&mb[b], f2o(s));
}
__global__ void cross_expsum_k(const float* __restrict__ sc, const unsigned* __restrict__ mb,
                               const int* __restrict__ batch, float* __restrict__ esc,
                               float* __restrict__ sb) {
  int n = blockIdx.x * blockDim.x + threadIdx.x;
  if (n >= NNODES) return;
  int b = batch[n];
  float ev = __expf(sc[n] - o2f(mb[b]));
  esc[n] = ev;
  atomicAdd(&sb[b], ev);
}
// attended = dn + attn*v[batch]; segment-max pool into ordered uints
__global__ void attend_pool_k(const float* __restrict__ dn, const float* __restrict__ vmat,
                              const float* __restrict__ esc, const float* __restrict__ sb,
                              const int* __restrict__ batch, unsigned* __restrict__ poolu) {
  int i = blockIdx.x * blockDim.x + threadIdx.x;
  if (i >= NNODES * 128) return;
  int n = i >> 7, c = i & 127;
  int b = batch[n];
  float attn = esc[n] / sb[b];
  float val = dn[(size_t)n * 128 + c] + attn * vmat[(size_t)b * 128 + c];
  atomicMax(&poolu[b * 128 + c], f2o(val));
}
// xc = [pooled(128, -inf->0), pv(256)]
__global__ void build_xc_k(const unsigned* __restrict__ poolu, const float* __restrict__ pv,
                           float* __restrict__ xc) {
  int i = blockIdx.x * blockDim.x + threadIdx.x;
  if (i >= NB * (DOUT + DP)) return;
  int b = i / (DOUT + DP), c = i - b * (DOUT + DP);
  float v;
  if (c < DOUT) {
    v = o2f(poolu[b * DOUT + c]);
    if (!(v >= -3.0e38f)) v = 0.0f;     // -inf / NaN -> 0
  } else {
    v = pv[(size_t)b * DP + (c - DOUT)];
  }
  xc[i] = v;
}

// final 512 -> 1 per graph
__global__ void final_dot_k(const float* __restrict__ h2, const float* __restrict__ w,
                            const float* __restrict__ b, float* __restrict__ out) {
  __shared__ float red[256];
  int g = blockIdx.x;
  float s = 0.f;
  for (int i = threadIdx.x; i < 512; i += 256) s += h2[(size_t)g * 512 + i] * w[i];
  red[threadIdx.x] = s;
  __syncthreads();
  for (int off = 128; off > 0; off >>= 1) {
    if (threadIdx.x < off) red[threadIdx.x] += red[threadIdx.x + off];
    __syncthreads();
  }
  if (threadIdx.x == 0) out[g] = red[0] + b[0];
}

// =====================================================================
// launch
// =====================================================================
static inline void gemm(const float* A, const float* W, const float* bias, float* C,
                        int M, int N, int K, int flags, hipStream_t st) {
  dim3 grid((N + 127) / 128, (M + 63) / 64);
  wmma_gemm_k<<<grid, 256, 0, st>>>(A, W, bias, C, M, N, K, flags);
}

extern "C" void kernel_launch(void* const* d_in, const int* in_sizes, int n_in,
                              void* d_out, int out_size, void* d_ws, size_t ws_size,
                              hipStream_t stream) {
  const float* x       = (const float*)d_in[0];
  const int*   ei      = (const int*)d_in[1];
  const int*   batch   = (const int*)d_in[2];
  const float* protein = (const float*)d_in[3];
  const float* pfc_w   = (const float*)d_in[4];
  const float* pfc_b   = (const float*)d_in[5];
  const float* gat_w   = (const float*)d_in[6];
  const float* att_src = (const float*)d_in[7];
  const float* att_dst = (const float*)d_in[8];
  const float* gat_b   = (const float*)d_in[9];
  const float* gcn_w   = (const float*)d_in[10];
  const float* gcn_b   = (const float*)d_in[11];
  const float* fcg1_w  = (const float*)d_in[12];
  const float* fcg1_b  = (const float*)d_in[13];
  const float* q_w     = (const float*)d_in[14];
  const float* q_b     = (const float*)d_in[15];
  const float* k_w     = (const float*)d_in[16];
  const float* k_b     = (const float*)d_in[17];
  const float* v_w     = (const float*)d_in[18];
  const float* v_b     = (const float*)d_in[19];
  const float* fc1_w   = (const float*)d_in[20];
  const float* fc1_b   = (const float*)d_in[21];
  const float* fc2_w   = (const float*)d_in[22];
  const float* fc2_b   = (const float*)d_in[23];
  const float* out_w   = (const float*)d_in[24];
  const float* out_b   = (const float*)d_in[25];
  float* out = (float*)d_out;
  float* ws  = (float*)d_ws;

  // workspace layout (floats)
  constexpr size_t O_PV    = 0;
  constexpr size_t O_HLIN  = O_PV    + (size_t)NB * DP;        // also reused for GCN linear
  constexpr size_t O_GOUT  = O_HLIN  + (size_t)NNODES * HF;    // GAT out; reused for GCN agg
  constexpr size_t O_ASRC  = O_GOUT  + (size_t)NNODES * HF;
  constexpr size_t O_ADST  = O_ASRC  + (size_t)NNODES * NH;
  constexpr size_t O_ALPHA = O_ADST  + (size_t)NNODES * NH;
  constexpr size_t O_MSEG  = O_ALPHA + (size_t)ETOT * NH;
  constexpr size_t O_SSEG  = O_MSEG  + (size_t)NNODES * NH;
  constexpr size_t O_DEG   = O_SSEG  + (size_t)NNODES * NH;
  constexpr size_t O_DINV  = O_DEG   + NNODES;
  constexpr size_t O_DN    = O_DINV  + NNODES;
  constexpr size_t O_Q     = O_DN    + (size_t)NNODES * DOUT;
  constexpr size_t O_K     = O_Q     + (size_t)NNODES * DOUT;
  constexpr size_t O_V     = O_K     + (size_t)NB * 128;
  constexpr size_t O_SC    = O_V     + (size_t)NB * 128;
  constexpr size_t O_ESC   = O_SC    + NNODES;
  constexpr size_t O_MB    = O_ESC   + NNODES;
  constexpr size_t O_SB    = O_MB    + NB;
  constexpr size_t O_POOL  = O_SB    + NB;
  constexpr size_t O_XC    = O_POOL  + (size_t)NB * DOUT;
  constexpr size_t O_H1    = O_XC    + (size_t)NB * (DOUT + DP);
  constexpr size_t O_H2    = O_H1    + (size_t)NB * 1024;
  constexpr size_t WS_END  = O_H2    + (size_t)NB * 512;
  if (ws_size < WS_END * sizeof(float)) return;

  float* PV   = ws + O_PV;
  float* HLIN = ws + O_HLIN;   // GAT linear out, later reused as GCN linear out
  float* GOUT = ws + O_GOUT;   // GAT aggregated (then relu'd h), later reused as GCN agg
  float* ASRC = ws + O_ASRC;
  float* ADST = ws + O_ADST;
  float* ALPH = ws + O_ALPHA;
  unsigned* MSEG = (unsigned*)(ws + O_MSEG);
  float* SSEG = ws + O_SSEG;
  float* DEG  = ws + O_DEG;
  float* DINV = ws + O_DINV;
  float* DN   = ws + O_DN;
  float* Q    = ws + O_Q;
  float* KM   = ws + O_K;
  float* VM   = ws + O_V;
  float* SC   = ws + O_SC;
  float* ESC  = ws + O_ESC;
  unsigned* MB = (unsigned*)(ws + O_MB);
  float* SB   = ws + O_SB;
  unsigned* POOLU = (unsigned*)(ws + O_POOL);
  float* XC   = ws + O_XC;
  float* H1   = ws + O_H1;
  float* H2   = ws + O_H2;

  auto blocks = [](int n) { return (n + 255) / 256; };

  // 1) protein fc: pv = relu(protein @ pfc_w + b)        [1024,256]
  gemm(protein, pfc_w, pfc_b, PV, NB, DP, DP, 3, stream);

  // 2) GAT linear: hlin = x @ gat_w                      [40000,780]
  gemm(x, gat_w, nullptr, HLIN, NNODES, HF, FXD, 0, stream);

  // 3) attention logits per (node, head)
  gat_scores_k<<<blocks(NNODES * NH), 256, 0, stream>>>(HLIN, att_src, att_dst, ASRC, ADST);

  // 4) segment softmax over dst + weighted aggregation
  fill_u32_k<<<blocks(NNODES * NH), 256, 0, stream>>>(MSEG, ORD_NEG_INF, NNODES * NH);
  fill_f32_k<<<blocks(NNODES * NH), 256, 0, stream>>>(SSEG, 0.f, NNODES * NH);
  fill_f32_k<<<blocks(NNODES * HF), 256, 0, stream>>>(GOUT, 0.f, NNODES * HF);
  edge_alpha_max_k<<<blocks(ETOT * NH), 256, 0, stream>>>(ei, ASRC, ADST, ALPH, MSEG);
  edge_expsum_k<<<blocks(ETOT * NH), 256, 0, stream>>>(ei, ALPH, MSEG, SSEG);
  gat_aggregate_k<<<ETOT, 256, 0, stream>>>(ei, ALPH, SSEG, HLIN, GOUT);
  bias_act_k<<<blocks(NNODES * HF), 256, 0, stream>>>(GOUT, gat_b, NNODES, HF, 1);

  // 5) GCN: degree norm, linear, aggregate
  fill_f32_k<<<blocks(NNODES), 256, 0, stream>>>(DEG, 0.f, NNODES);
  deg_k<<<blocks(ETOT), 256, 0, stream>>>(ei, DEG);
  dinv_k<<<blocks(NNODES), 256, 0, stream>>>(DEG, DINV);
  gemm(GOUT, gcn_w, nullptr, HLIN, NNODES, HF, HF, 0, stream);   // hW -> reuse HLIN
  fill_f32_k<<<blocks(NNODES * HF), 256, 0, stream>>>(GOUT, 0.f, NNODES * HF); // reuse GOUT as agg
  gcn_aggregate_k<<<ETOT, 256, 0, stream>>>(ei, DINV, HLIN, GOUT);
  bias_act_k<<<blocks(NNODES * HF), 256, 0, stream>>>(GOUT, gcn_b, NNODES, HF, 1);

  // 6) fc_g1: dn = relu(drug_nodes @ fcg1_w + b)         [40000,128]
  gemm(GOUT, fcg1_w, fcg1_b, DN, NNODES, DOUT, HF, 3, stream);

  // 7) cross attention projections
  gemm(DN, q_w, q_b, Q, NNODES, 128, DOUT, 2, stream);
  gemm(PV, k_w, k_b, KM, NB, 128, DP, 2, stream);
  gemm(PV, v_w, v_b, VM, NB, 128, DP, 2, stream);

  // 8) per-graph softmax over nodes + attended + max pool
  fill_u32_k<<<blocks(NB), 256, 0, stream>>>(MB, ORD_NEG_INF, NB);
  fill_f32_k<<<blocks(NB), 256, 0, stream>>>(SB, 0.f, NB);
  fill_u32_k<<<blocks(NB * DOUT), 256, 0, stream>>>(POOLU, ORD_NEG_INF, NB * DOUT);
  cross_scores_k<<<blocks(NNODES), 256, 0, stream>>>(Q, KM, batch, SC, MB);
  cross_expsum_k<<<blocks(NNODES), 256, 0, stream>>>(SC, MB, batch, ESC, SB);
  attend_pool_k<<<blocks(NNODES * 128), 256, 0, stream>>>(DN, VM, ESC, SB, batch, POOLU);
  build_xc_k<<<blocks(NB * (DOUT + DP)), 256, 0, stream>>>(POOLU, PV, XC);

  // 9) head MLP
  gemm(XC, fc1_w, fc1_b, H1, NB, 1024, DOUT + DP, 3, stream);
  gemm(H1, fc2_w, fc2_b, H2, NB, 512, 1024, 3, stream);
  final_dot_k<<<NB, 256, 0, stream>>>(H2, out_w, out_b, out);
}
